// TransformerBlock_52089363366220
// MI455X (gfx1250) — matrix operand order repaired and checked by reading, not compiled, not run on hardware
//
#include <hip/hip_runtime.h>

// ---------------------------------------------------------------------------
// Types for CDNA5 WMMA (wave32): v16bf = 16 bf16 (8 VGPRs), v8f = 8 f32
// ---------------------------------------------------------------------------
typedef __attribute__((ext_vector_type(16))) __bf16 v16bf;
typedef __attribute__((ext_vector_type(8)))  float  v8f;
typedef __attribute__((ext_vector_type(8)))  unsigned int v8u;

union FragCvt {
  v8u u;
  v16bf b;
};

// ---------------------------------------------------------------------------
// Fragment loaders, CDNA5 ISA 7.12.2 layouts (wave32), on u32-typed LDS so
// each fragment is two aligned 16-byte LDS loads (no bf16 packing VALU).
//
// A (16x32 bf16), row-major tile, ld2 = row stride in u32:
//   lane m=lane&15, h=lane>>4; u32 slot v<4 : k2 = col2 + 4h + v
//                              u32 slot v>=4: k2 = col2 + 8 + 4h + (v-4)
// B (32x16 bf16) gathered from an n-major tile T[n][k] (k-pairs contiguous):
//   lane n=lane&15, h=lane>>4; u32 slot v<4 : k2 = k2_0 + 4h + v
//                              u32 slot v>=4: k2 = k2_0 + 8 + 4h + (v-4)
// ---------------------------------------------------------------------------
__device__ __forceinline__ v16bf load_frag_a32(const unsigned int* lds,
                                               int ld2, int row0, int col2,
                                               int lane) {
  const int m = lane & 15, h = lane >> 4;
  const unsigned int* p = lds + (row0 + m) * ld2 + col2;
  const uint4 lo = *(const uint4*)(p + 4 * h);
  const uint4 hi = *(const uint4*)(p + 8 + 4 * h);
  FragCvt c;
  c.u[0] = lo.x; c.u[1] = lo.y; c.u[2] = lo.z; c.u[3] = lo.w;
  c.u[4] = hi.x; c.u[5] = hi.y; c.u[6] = hi.z; c.u[7] = hi.w;
  return c.b;
}

__device__ __forceinline__ v16bf load_frag_bT32(const unsigned int* lds,
                                                int ld2, int k2_0, int ncol0,
                                                int lane) {
  const int n = lane & 15, h = lane >> 4;
  const unsigned int* p = lds + (ncol0 + n) * ld2 + k2_0;
  const uint4 lo = *(const uint4*)(p + 4 * h);
  const uint4 hi = *(const uint4*)(p + 8 + 4 * h);
  FragCvt c;
  c.u[0] = lo.x; c.u[1] = lo.y; c.u[2] = lo.z; c.u[3] = lo.w;
  c.u[4] = hi.x; c.u[5] = hi.y; c.u[6] = hi.z; c.u[7] = hi.w;
  return c.b;
}

// ---------------------------------------------------------------------------
// fp32 -> bf16 conversion (plain, and zero-padding variant for LoRA factors)
// ---------------------------------------------------------------------------
__global__ void cvt_f32_to_bf16_kernel(const float* __restrict__ src,
                                       __bf16* __restrict__ dst, int n) {
  const int i = blockIdx.x * blockDim.x + threadIdx.x;
  if (i < n) dst[i] = (__bf16)src[i];
}

__global__ void cvt_pad_bf16_kernel(const float* __restrict__ src, int R,
                                    int C, __bf16* __restrict__ dst, int Rp,
                                    int Cp) {
  const int i = blockIdx.x * blockDim.x + threadIdx.x;
  if (i >= Rp * Cp) return;
  const int r = i / Cp, c = i % Cp;
  dst[i] = (r < R && c < C) ? (__bf16)src[r * C + c] : (__bf16)0.0f;
}

// ---------------------------------------------------------------------------
// Row LayerNorm (fp32 stats) emitting bf16 activations. One block per row.
// ---------------------------------------------------------------------------
__global__ __launch_bounds__(256) void layernorm_bf16_kernel(
    const float* __restrict__ x, const float* __restrict__ g,
    const float* __restrict__ bta, __bf16* __restrict__ out, int C) {
  __shared__ float red[256];
  const int row = blockIdx.x;
  const float* xr = x + (size_t)row * C;
  float s = 0.f;
  for (int c = threadIdx.x; c < C; c += 256) s += xr[c];
  red[threadIdx.x] = s;
  __syncthreads();
  for (int off = 128; off > 0; off >>= 1) {
    if (threadIdx.x < off) red[threadIdx.x] += red[threadIdx.x + off];
    __syncthreads();
  }
  const float mu = red[0] / C;
  __syncthreads();
  float vv = 0.f;
  for (int c = threadIdx.x; c < C; c += 256) {
    const float d = xr[c] - mu;
    vv += d * d;
  }
  red[threadIdx.x] = vv;
  __syncthreads();
  for (int off = 128; off > 0; off >>= 1) {
    if (threadIdx.x < off) red[threadIdx.x] += red[threadIdx.x + off];
    __syncthreads();
  }
  const float rstd = rsqrtf(red[0] / C + 1e-5f);
  for (int c = threadIdx.x; c < C; c += 256)
    out[(size_t)row * C + c] = (__bf16)((xr[c] - mu) * rstd * g[c] + bta[c]);
}

// ---------------------------------------------------------------------------
// bf16 WMMA GEMM, 64x128 tile per 256-thread block (8 waves; each wave owns a
// 16-row M tile x four 16-col N tiles -> 1 A-frag feeds 4 WMMAs per K-step).
// Requires M%64==0, N%128==0, K%32==0 (guaranteed by LoRA padding).
// Epilogue (compile-time specialized): alpha*A@B [+bias] [+Cin] [ReLU]
// -> fp32 and/or bf16 outputs.
// ---------------------------------------------------------------------------
template <bool HAS_BIAS, bool HAS_CIN, bool HAS_COUT, bool HAS_BOUT, bool RELU>
__global__ __launch_bounds__(256) void gemm_bf16_kernel(
    const __bf16* __restrict__ A, int lda, const __bf16* __restrict__ B,
    int ldb, const float* __restrict__ bias, const float* __restrict__ Cin,
    float* __restrict__ Cout, __bf16* __restrict__ Bout, int M, int N, int K,
    float alpha) {
  __shared__ __align__(16) __bf16 As[64 * 32];   // [m][k] row-major
  __shared__ __align__(16) __bf16 Bt[128 * 32];  // [n][k] (transposed)
  unsigned int* As32 = (unsigned int*)As;
  unsigned int* Bt32 = (unsigned int*)Bt;
  const int tid = threadIdx.x, lane = tid & 31, w = tid >> 5;
  const int mt = w & 3;
  const int ntb = (w >> 2) * 4;  // 0 or 4
  const int m0 = blockIdx.y * 64, n0 = blockIdx.x * 128;
  v8f acc[4] = {};

  for (int k0 = 0; k0 < K; k0 += 32) {
    __syncthreads();
    {  // stage A 64x32: one b128 global load + one b128 LDS store per thread
      const int ar = tid >> 2, ac = (tid & 3) * 8;
      const uint4 av = *(const uint4*)(A + (size_t)(m0 + ar) * lda + k0 + ac);
      *(uint4*)(As + ar * 32 + ac) = av;
    }
    {  // stage B 32x128 transposed: pack K-pairs into b32 LDS stores
#pragma unroll
      for (int s = 0; s < 2; ++s) {
        const int idx = tid + s * 256;
        const int kp = idx >> 5;        // 0..15 (pair of k rows)
        const int nb = (idx & 31) * 4;  // 0..124
        const __bf16* bp = B + (size_t)(k0 + 2 * kp) * ldb + n0 + nb;
        const uint2 r0 = *(const uint2*)bp;
        const uint2 r1 = *(const uint2*)(bp + ldb);
        unsigned int* d = Bt32 + nb * 16 + kp;
        d[0]  = (r0.x & 0xffffu) | (r1.x << 16);
        d[16] = (r0.x >> 16) | (r1.x & 0xffff0000u);
        d[32] = (r0.y & 0xffffu) | (r1.y << 16);
        d[48] = (r0.y >> 16) | (r1.y & 0xffff0000u);
      }
    }
    __syncthreads();
    const v16bf a = load_frag_a32(As32, 16, 16 * mt, 0, lane);
#pragma unroll
    for (int t = 0; t < 4; ++t) {
      const v16bf b = load_frag_bT32(Bt32, 16, 0, 16 * (ntb + t), lane);
      acc[t] = __builtin_amdgcn_wmma_f32_16x16x32_bf16(
          false, a, false, b, (short)0, acc[t], false, false);
    }
  }

  const int h = lane >> 4, nn = lane & 15;
#pragma unroll
  for (int t = 0; t < 4; ++t) {
    const int col = n0 + 16 * (ntb + t) + nn;
    float bsv = 0.0f;
    if (HAS_BIAS) bsv = bias[col];
#pragma unroll
    for (int r = 0; r < 8; ++r) {
      const int row = m0 + 16 * mt + r + 8 * h;
      float val = alpha * acc[t][r] + bsv;
      if (HAS_CIN) val += Cin[(size_t)row * N + col];
      if (RELU) val = fmaxf(val, 0.0f);
      if (HAS_COUT) Cout[(size_t)row * N + col] = val;
      if (HAS_BOUT) Bout[(size_t)row * N + col] = (__bf16)val;
    }
  }
}

// ---------------------------------------------------------------------------
// Flash attention (GQA, causal). Block = 64 query rows of one (batch, head).
// Q@K^T and P@V on WMMA; online softmax with 4-way-parallel row pass.
// ---------------------------------------------------------------------------
__global__ __launch_bounds__(256) void attn_kernel(
    const __bf16* __restrict__ Qb, const __bf16* __restrict__ Kb,
    const __bf16* __restrict__ Vb, __bf16* __restrict__ Ob, int T) {
  __shared__ __align__(16) __bf16 Qs[64 * 64];  // [q][d]
  __shared__ __align__(16) __bf16 Ks[64 * 64];  // [key][d]
  __shared__ __align__(16) __bf16 Vt[64 * 64];  // [d][key] (transposed)
  __shared__ __align__(16) __bf16 Ps[64 * 64];  // [q][key]
  __shared__ float Ss[64 * 64];
  __shared__ float pm[256], ps[256];
  __shared__ float mstate[64], lstate[64], cstate[64];
  unsigned int* Qs32 = (unsigned int*)Qs;
  unsigned int* Ks32 = (unsigned int*)Ks;
  unsigned int* Vt32 = (unsigned int*)Vt;
  unsigned int* Ps32 = (unsigned int*)Ps;

  const int tid = threadIdx.x, lane = tid & 31, w = tid >> 5;
  const int mt = w & 3, nt0 = (w >> 2) * 2;
  const int q0 = blockIdx.x * 64;
  const int bh = blockIdx.y;
  const int bb = bh >> 4, hq = bh & 15;
  const int grp = hq >> 2;  // GQA: 4 q-heads per kv group
  const size_t qbase = ((size_t)bb * T) * 1024 + (size_t)hq * 64;
  const size_t kbase = ((size_t)bb * T) * 256 + (size_t)grp * 64;

  for (int s = tid; s < 512; s += 256) {  // Q tile, b128 copies
    const int r = s >> 3, c = (s & 7) * 8;
    *(uint4*)(Qs + r * 64 + c) =
        *(const uint4*)(Qb + qbase + (size_t)(q0 + r) * 1024 + c);
  }
  if (tid < 64) {
    mstate[tid] = -1e30f;
    lstate[tid] = 0.0f;
  }
  v8f o0 = {}, o1 = {};

  for (int j0 = 0; j0 <= q0; j0 += 64) {
    __syncthreads();
    for (int s = tid; s < 512; s += 256) {  // K tile row-major, b128 copies
      const int r = s >> 3, c = (s & 7) * 8;
      *(uint4*)(Ks + r * 64 + c) =
          *(const uint4*)(Kb + kbase + (size_t)(j0 + r) * 256 + c);
    }
#pragma unroll
    for (int s = 0; s < 2; ++s) {  // V tile transposed, packed b32 stores
      const int idx = tid + s * 256;
      const int kp = idx >> 4;        // 0..31 (key pair)
      const int nb = (idx & 15) * 4;  // 0..60 (dim)
      const __bf16* vp = Vb + kbase + (size_t)(j0 + 2 * kp) * 256 + nb;
      const uint2 r0 = *(const uint2*)vp;
      const uint2 r1 = *(const uint2*)(vp + 256);
      unsigned int* d = Vt32 + nb * 32 + kp;
      d[0]  = (r0.x & 0xffffu) | (r1.x << 16);
      d[32] = (r0.x >> 16) | (r1.x & 0xffff0000u);
      d[64] = (r0.y & 0xffffu) | (r1.y << 16);
      d[96] = (r0.y >> 16) | (r1.y & 0xffff0000u);
    }
    __syncthreads();

    // S = Q @ K^T (contraction over head_dim 64 -> two K32 steps); the
    // transposed-B gather from row-major Ks is naturally pair-contiguous.
    v8f s0 = {}, s1 = {};
#pragma unroll
    for (int kk = 0; kk < 64; kk += 32) {
      const v16bf a  = load_frag_a32(Qs32, 32, 16 * mt, kk / 2, lane);
      const v16bf b0 = load_frag_bT32(Ks32, 32, kk / 2, 16 * nt0, lane);
      const v16bf b1 = load_frag_bT32(Ks32, 32, kk / 2, 16 * (nt0 + 1), lane);
      s0 = __builtin_amdgcn_wmma_f32_16x16x32_bf16(false, a, false, b0,
                                                   (short)0, s0, false, false);
      s1 = __builtin_amdgcn_wmma_f32_16x16x32_bf16(false, a, false, b1,
                                                   (short)0, s1, false, false);
    }
    {  // scale + causal mask, spill S tiles to LDS
      const int h = lane >> 4, n = lane & 15;
#pragma unroll
      for (int r = 0; r < 8; ++r) {
        const int rl = 16 * mt + r + 8 * h;
        const int rg = q0 + rl;
#pragma unroll
        for (int t = 0; t < 2; ++t) {
          const int cl = 16 * (nt0 + t) + n;
          float vS = (t == 0 ? s0[r] : s1[r]) * 0.125f;  // 64^-0.5
          if (j0 + cl > rg) vS = -3.0e38f;
          Ss[rl * 64 + cl] = vS;
        }
      }
    }
    __syncthreads();
    // ---- online softmax, 4 threads per row ----
    const int rrow = tid & 63, rq = tid >> 6;
    {
      float mx = -3.4e38f;
      const float* Sr = Ss + rrow * 64 + rq * 16;
#pragma unroll
      for (int c = 0; c < 16; ++c) mx = fmaxf(mx, Sr[c]);
      pm[tid] = mx;
    }
    __syncthreads();
    if (tid < 64) {
      const float mo = mstate[tid];
      float mx = fmaxf(fmaxf(pm[tid], pm[tid + 64]),
                       fmaxf(pm[tid + 128], pm[tid + 192]));
      mx = fmaxf(mx, mo);
      mstate[tid] = mx;
      cstate[tid] = __expf(mo - mx);
    }
    __syncthreads();
    {
      const float mx = mstate[rrow];
      const float* Sr = Ss + rrow * 64 + rq * 16;
      __bf16* Pr = Ps + rrow * 64 + rq * 16;
      float sum = 0.0f;
#pragma unroll
      for (int c = 0; c < 16; ++c) {
        const float p = __expf(Sr[c] - mx);
        Pr[c] = (__bf16)p;
        sum += p;
      }
      ps[tid] = sum;
    }
    __syncthreads();
    if (tid < 64)
      lstate[tid] = lstate[tid] * cstate[tid] + ps[tid] + ps[tid + 64] +
                    ps[tid + 128] + ps[tid + 192];
    __syncthreads();
    {  // rescale running O
      const int h = lane >> 4;
#pragma unroll
      for (int r = 0; r < 8; ++r) {
        const float corr = cstate[16 * mt + r + 8 * h];
        o0[r] *= corr;
        o1[r] *= corr;
      }
    }
#pragma unroll
    for (int kk = 0; kk < 64; kk += 32) {  // O += P @ V
      const v16bf a  = load_frag_a32(Ps32, 32, 16 * mt, kk / 2, lane);
      const v16bf b0 = load_frag_bT32(Vt32, 32, kk / 2, 16 * nt0, lane);
      const v16bf b1 = load_frag_bT32(Vt32, 32, kk / 2, 16 * (nt0 + 1), lane);
      o0 = __builtin_amdgcn_wmma_f32_16x16x32_bf16(false, a, false, b0,
                                                   (short)0, o0, false, false);
      o1 = __builtin_amdgcn_wmma_f32_16x16x32_bf16(false, a, false, b1,
                                                   (short)0, o1, false, false);
    }
  }
  {  // normalize and write bf16 output
    const int h = lane >> 4, n = lane & 15;
#pragma unroll
    for (int r = 0; r < 8; ++r) {
      const int rl = 16 * mt + r + 8 * h;
      const float inv = 1.0f / lstate[rl];
      const size_t rowoff = qbase + (size_t)(q0 + rl) * 1024;
      Ob[rowoff + 16 * nt0 + n]       = (__bf16)(o0[r] * inv);
      Ob[rowoff + 16 * (nt0 + 1) + n] = (__bf16)(o1[r] * inv);
    }
  }
}

// ---------------------------------------------------------------------------
// Host orchestration
// ---------------------------------------------------------------------------
static inline int cdiv(int a, int b) { return (a + b - 1) / b; }

template <bool HB, bool HC, bool CO, bool BO, bool RL>
static void launch_gemm(const __bf16* A, int lda, const __bf16* B, int ldb,
                        const float* bias, const float* Cin, float* Cout,
                        __bf16* Bout, int M, int N, int K, float alpha,
                        hipStream_t stream) {
  dim3 grid(N / 128, M / 64);
  gemm_bf16_kernel<HB, HC, CO, BO, RL><<<grid, 256, 0, stream>>>(
      A, lda, B, ldb, bias, Cin, Cout, Bout, M, N, K, alpha);
}

extern "C" void kernel_launch(void* const* d_in, const int* in_sizes, int n_in,
                              void* d_out, int out_size, void* d_ws,
                              size_t ws_size, hipStream_t stream) {
  (void)in_sizes; (void)n_in; (void)out_size; (void)ws_size;
  const int Bsz = 2, T = 2048, D = 1024, KV = 256, R = 16, E = 4096;
  const int Rp = 128;        // LoRA rank padded to GEMM tile (zeros)
  const int Mtok = Bsz * T;  // 4096 token rows
  const float LSCALE = 2.0f; // LORA_ALPHA / LORA_RANK = 32/16

  const float* x   = (const float*)d_in[0];
  const float* Wq  = (const float*)d_in[2];
  const float* bq  = (const float*)d_in[3];
  const float* Wk  = (const float*)d_in[4];
  const float* bk  = (const float*)d_in[5];
  const float* Wv  = (const float*)d_in[6];
  const float* bv  = (const float*)d_in[7];
  const float* Wo  = (const float*)d_in[8];
  const float* bo  = (const float*)d_in[9];
  const float* lqa = (const float*)d_in[10];
  const float* lqb = (const float*)d_in[11];
  const float* lva = (const float*)d_in[12];
  const float* lvb = (const float*)d_in[13];
  const float* W1  = (const float*)d_in[14];
  const float* b1  = (const float*)d_in[15];
  const float* W2  = (const float*)d_in[16];
  const float* b2  = (const float*)d_in[17];
  const float* g1  = (const float*)d_in[18];
  const float* be1 = (const float*)d_in[19];
  const float* g2  = (const float*)d_in[20];
  const float* be2 = (const float*)d_in[21];

  char* ws = (char*)d_ws;
  size_t off = 0;
  auto alloc = [&](size_t bytes) -> void* {
    void* p = ws + off;
    off += (bytes + 255) & ~(size_t)255;
    return p;
  };

  __bf16* Wq_b  = (__bf16*)alloc((size_t)D * D * 2);
  __bf16* Wk_b  = (__bf16*)alloc((size_t)D * KV * 2);
  __bf16* Wv_b  = (__bf16*)alloc((size_t)D * KV * 2);
  __bf16* Wo_b  = (__bf16*)alloc((size_t)D * D * 2);
  __bf16* W1_b  = (__bf16*)alloc((size_t)D * E * 2);
  __bf16* W2_b  = (__bf16*)alloc((size_t)E * D * 2);
  __bf16* lqa_b = (__bf16*)alloc((size_t)D * Rp * 2);
  __bf16* lqb_b = (__bf16*)alloc((size_t)Rp * D * 2);
  __bf16* lva_b = (__bf16*)alloc((size_t)D * Rp * 2);
  __bf16* lvb_b = (__bf16*)alloc((size_t)Rp * KV * 2);
  __bf16* h_b   = (__bf16*)alloc((size_t)Mtok * D * 2);
  float*  q_f   = (float*)alloc((size_t)Mtok * D * 4);
  __bf16* tq_b  = (__bf16*)alloc((size_t)Mtok * Rp * 2);
  __bf16* q_b   = (__bf16*)alloc((size_t)Mtok * D * 2);
  __bf16* k_b   = (__bf16*)alloc((size_t)Mtok * KV * 2);
  float*  v_f   = (float*)alloc((size_t)Mtok * KV * 4);
  __bf16* tv_b  = (__bf16*)alloc((size_t)Mtok * Rp * 2);
  __bf16* v_b   = (__bf16*)alloc((size_t)Mtok * KV * 2);
  __bf16* at_b  = (__bf16*)alloc((size_t)Mtok * D * 2);
  float*  x2_f  = (float*)alloc((size_t)Mtok * D * 4);
  __bf16* h2_b  = (__bf16*)alloc((size_t)Mtok * D * 2);
  __bf16* u_b   = (__bf16*)alloc((size_t)Mtok * E * 2);

  auto cvt = [&](const float* s, __bf16* d, int n) {
    cvt_f32_to_bf16_kernel<<<cdiv(n, 256), 256, 0, stream>>>(s, d, n);
  };
  cvt(Wq, Wq_b, D * D);
  cvt(Wk, Wk_b, D * KV);
  cvt(Wv, Wv_b, D * KV);
  cvt(Wo, Wo_b, D * D);
  cvt(W1, W1_b, D * E);
  cvt(W2, W2_b, E * D);
  auto cvtpad = [&](const float* s, int Rr, int Cc, __bf16* d, int Rpp,
                    int Cpp) {
    cvt_pad_bf16_kernel<<<cdiv(Rpp * Cpp, 256), 256, 0, stream>>>(s, Rr, Cc, d,
                                                                  Rpp, Cpp);
  };
  cvtpad(lqa, D, R, lqa_b, D, Rp);  // 1024x16 -> 1024x128 (cols padded)
  cvtpad(lqb, R, D, lqb_b, Rp, D);  // 16x1024 -> 128x1024 (rows padded)
  cvtpad(lva, D, R, lva_b, D, Rp);
  cvtpad(lvb, R, KV, lvb_b, Rp, KV);

  // h = LN1(x) in bf16
  layernorm_bf16_kernel<<<Mtok, 256, 0, stream>>>(x, g1, be1, h_b, D);

  // q = h@Wq + bq  (+ LoRA, rank padded to 128 with zeros)
  launch_gemm<true, false, true, false, false>(h_b, D, Wq_b, D, bq, nullptr,
                                               q_f, nullptr, Mtok, D, D, 1.0f,
                                               stream);
  launch_gemm<false, false, false, true, false>(h_b, D, lqa_b, Rp, nullptr,
                                                nullptr, nullptr, tq_b, Mtok,
                                                Rp, D, 1.0f, stream);
  launch_gemm<false, true, false, true, false>(tq_b, Rp, lqb_b, D, nullptr,
                                               q_f, nullptr, q_b, Mtok, D, Rp,
                                               LSCALE, stream);
  // k = h@Wk + bk
  launch_gemm<true, false, false, true, false>(h_b, D, Wk_b, KV, bk, nullptr,
                                               nullptr, k_b, Mtok, KV, D, 1.0f,
                                               stream);
  // v = h@Wv + bv (+ LoRA)
  launch_gemm<true, false, true, false, false>(h_b, D, Wv_b, KV, bv, nullptr,
                                               v_f, nullptr, Mtok, KV, D, 1.0f,
                                               stream);
  launch_gemm<false, false, false, true, false>(h_b, D, lva_b, Rp, nullptr,
                                                nullptr, nullptr, tv_b, Mtok,
                                                Rp, D, 1.0f, stream);
  launch_gemm<false, true, false, true, false>(tv_b, Rp, lvb_b, KV, nullptr,
                                               v_f, nullptr, v_b, Mtok, KV, Rp,
                                               LSCALE, stream);

  // causal GQA flash attention -> at_b (bf16, token-major [B*T, 16*64])
  {
    dim3 grid(T / 64, Bsz * 16);
    attn_kernel<<<grid, 256, 0, stream>>>(q_b, k_b, v_b, at_b, T);
  }

  // x2 = x + attn@Wo + bo
  launch_gemm<true, true, true, false, false>(at_b, D, Wo_b, D, bo, x, x2_f,
                                              nullptr, Mtok, D, D, 1.0f,
                                              stream);
  // h2 = LN2(x2)
  layernorm_bf16_kernel<<<Mtok, 256, 0, stream>>>(x2_f, g2, be2, h2_b, D);
  // u = relu(h2@W1 + b1)
  launch_gemm<true, false, false, true, true>(h2_b, D, W1_b, E, b1, nullptr,
                                              nullptr, u_b, Mtok, E, D, 1.0f,
                                              stream);
  // out = x2 + u@W2 + b2
  launch_gemm<true, true, true, false, false>(u_b, E, W2_b, D, b2, x2_f,
                                              (float*)d_out, nullptr, Mtok, D,
                                              E, 1.0f, stream);
}
